// CVAE_64098091926117
// MI455X (gfx1250) — matrix operand order repaired and checked by reading, not compiled, not run on hardware
//
#include <hip/hip_runtime.h>
#include <hip/hip_bf16.h>

// ---------------- problem constants ----------------
#define Bdim   16
#define Lm1    1023
#define Hdim   1024
#define Gdim   128
#define Mrows  (Bdim * Lm1)       // 16368
#define Ktot   3072
#define Kz     2048
#define KC     32                 // K chunk per stage
#define KP     (KC / 2)           // packed bf16 pairs per chunk-row = 16
#define APAD   20                 // padded dword stride (16B-aligned, conflict-avoiding)
#define WPAD   20
#define MTILE  64
#define NTHREADS 512

typedef __attribute__((ext_vector_type(16))) __bf16 v16bf;
typedef __attribute__((ext_vector_type(8)))  float  v8f;

union Frag { unsigned int u[8]; v16bf v; };

__device__ __forceinline__ unsigned int pack2_bf16(float a, float b) {
  // round-to-nearest-even f32 -> bf16, two values packed into one dword
  unsigned int ua = __float_as_uint(a);
  unsigned int ub = __float_as_uint(b);
  ua += 0x7fffu + ((ua >> 16) & 1u);
  ub += 0x7fffu + ((ub >> 16) & 1u);
  return (ua >> 16) | (ub & 0xffff0000u);
}

// chained ternary over raw kernel args (keeps global addrspace inference alive)
__device__ __forceinline__ const float* sel_w(int mat, const float* w0,
                                              const float* w1, const float* w2,
                                              const float* w3) {
  const float* r = w3;
  r = (mat == 2) ? w2 : r;
  r = (mat == 1) ? w1 : r;
  r = (mat == 0) ? w0 : r;
  return r;
}

// Register-staged prefetch payload for one K chunk (per thread).
struct Pref {
  float2 a[2];                // 2 A pair-elements
  float4 w0[4], w1[4];        // up to 4 weight groups (4 cols x 2 k-rows)
};

// W-staging work decomposition: matrix index is WAVE-UNIFORM so the global
// base pointer can live in SGPRs (global_load saddr form).
__device__ __forceinline__ void wstage_map(int k0, int tid, int i,
                                           int& mat, int& cg, int& kp, int& iters) {
  const int lane  = tid & 31;
  const int wid   = tid >> 5;
  const int shift = (k0 < Kz) ? 2 : 3;          // log2(waves per matrix)
  const int mbase = (k0 < Kz) ? 0 : 2;
  mat   = mbase + (wid >> shift);
  iters = 512 >> (shift + 5);                   // 4 (nmat=4) or 2 (nmat=2)
  const int wim   = wid & ((1 << shift) - 1);
  const int slots = 32 << shift;                // lanes covering one matrix
  const int g     = wim * 32 + lane + i * slots;  // [0, 512) per matrix
  cg = g & 31;                                  // 4-col group
  kp = g >> 5;                                  // packed-k index
}

__device__ __forceinline__ void load_chunk_regs(
    int k0, int m0, int tid,
    const float* __restrict__ events, const float* __restrict__ contexts,
    const float* __restrict__ Wzm, const float* __restrict__ Wzv,
    const float* __restrict__ Wqm, const float* __restrict__ Wqv, Pref& p) {
  // ---- A: virtual row m over K = [events_l | events_l+1 | contexts_l]
#pragma unroll
  for (int i = 0; i < 2; ++i) {
    int e  = tid + i * NTHREADS;
    int r  = e >> 4;
    int kp = e & (KP - 1);
    unsigned int m = (unsigned int)(m0 + r);
    float2 v = {0.f, 0.f};
    if (m < (unsigned int)Mrows) {
      unsigned int b = m / 1023u;  // batch index; events row = m + b
      if (k0 < Kz) {
        const float* src = events + (size_t)(m + b) * Hdim
                         + (size_t)(k0 & (Hdim - 1))
                         + ((k0 >= Hdim) ? (size_t)Hdim : 0);
        v = *(const float2*)(src + 2 * kp);
      } else {
        const float* src = contexts + (size_t)m * Hdim + (size_t)(k0 - Kz);
        v = *(const float2*)(src + 2 * kp);
      }
    }
    p.a[i] = v;
  }
  // ---- W: float4 row segments (coalesced, wave-uniform matrix)
#pragma unroll
  for (int i = 0; i < 4; ++i) {
    int mat, cg, kp, iters;
    wstage_map(k0, tid, i, mat, cg, kp, iters);
    if (i < iters) {
      const float* w = sel_w(mat, Wzm, Wzv, Wqm, Wqv);
      size_t off = (size_t)(k0 + 2 * kp) * Gdim + cg * 4;
      p.w0[i] = *(const float4*)(w + off);
      p.w1[i] = *(const float4*)(w + off + Gdim);
      // speculative prefetch of this weight stream two chunks ahead
      if (k0 + 2 * KC < ((mat < 2) ? Kz : Ktot))
        __builtin_prefetch(w + off + (size_t)(2 * KC) * Gdim, 0, 1);
    }
  }
}

__device__ __forceinline__ void store_chunk_lds(
    int k0, int tid, const Pref& p,
    unsigned int* __restrict__ sA, unsigned int* __restrict__ sW) {
#pragma unroll
  for (int i = 0; i < 2; ++i) {
    int e  = tid + i * NTHREADS;
    int r  = e >> 4;
    int kp = e & (KP - 1);
    sA[r * APAD + kp] = pack2_bf16(p.a[i].x, p.a[i].y);
  }
#pragma unroll
  for (int i = 0; i < 4; ++i) {
    int mat, cg, kp, iters;
    wstage_map(k0, tid, i, mat, cg, kp, iters);
    if (i < iters) {
      unsigned int* base = sW + (size_t)mat * (Gdim * WPAD);
      base[(cg * 4 + 0) * WPAD + kp] = pack2_bf16(p.w0[i].x, p.w1[i].x);
      base[(cg * 4 + 1) * WPAD + kp] = pack2_bf16(p.w0[i].y, p.w1[i].y);
      base[(cg * 4 + 2) * WPAD + kp] = pack2_bf16(p.w0[i].z, p.w1[i].z);
      base[(cg * 4 + 3) * WPAD + kp] = pack2_bf16(p.w0[i].w, p.w1[i].w);
    }
  }
}

// One workgroup: 64 M-rows x 128 cols x 4 output heads.
// 16 waves: wave = (out_id 0..3) x (16-row m-subtile 0..3).
__global__ __launch_bounds__(NTHREADS) void cvae_gemm_kernel(
    const float* __restrict__ events,  const float* __restrict__ contexts,
    const float* __restrict__ Wzm, const float* __restrict__ bzm,
    const float* __restrict__ Wzv, const float* __restrict__ bzv,
    const float* __restrict__ Wqm, const float* __restrict__ bqm,
    const float* __restrict__ Wqv, const float* __restrict__ bqv,
    float* __restrict__ out) {
  __shared__ unsigned int sA[MTILE * APAD];        //  5 KB
  __shared__ unsigned int sW[4][Gdim * WPAD];      // 40 KB

  const int tid    = threadIdx.x;
  const int lane   = tid & 31;
  const int wid    = tid >> 5;
  const int out_id = wid >> 2;       // 0:z_mean 1:z_log_var 2:q_mean 3:q_log_var
  const int ms     = wid & 3;        // 16-row m-subtile
  const int half   = lane >> 4;      // K-half selector per ISA fragment layout
  const int l15    = lane & 15;
  const int m0     = blockIdx.x * MTILE;

  v8f acc[8];
#pragma unroll
  for (int i = 0; i < 8; ++i) {
    v8f z = {0.f, 0.f, 0.f, 0.f, 0.f, 0.f, 0.f, 0.f};
    acc[i] = z;
  }

  Pref pf;
  load_chunk_regs(0, m0, tid, events, contexts, Wzm, Wzv, Wqm, Wqv, pf);

  for (int k0 = 0; k0 < Ktot; k0 += KC) {
    __syncthreads();                       // previous chunk's LDS consumers done
    store_chunk_lds(k0, tid, pf, sA, &sW[0][0]);
    __syncthreads();

    // software pipeline: prefetch next chunk's global data into registers;
    // loadcnt waits land at the *next* iteration's LDS-store phase.
    if (k0 + KC < Ktot)
      load_chunk_regs(k0 + KC, m0, tid, events, contexts, Wzm, Wzv, Wqm, Wqv, pf);

    if (k0 >= Kz && out_id < 2) continue;  // z-heads only use K < 2048

    // A fragment (ISA 16-bit A 16x32: lane<16 K{0..7,16..23}; lane>=16 K{8..15,24..31})
    Frag af;
    {
      const unsigned int* pa = &sA[(ms * 16 + l15) * APAD + half * 4];
      uint4 a0 = *(const uint4*)pa;
      uint4 a1 = *(const uint4*)(pa + 8);
      af.u[0] = a0.x; af.u[1] = a0.y; af.u[2] = a0.z; af.u[3] = a0.w;
      af.u[4] = a1.x; af.u[5] = a1.y; af.u[6] = a1.z; af.u[7] = a1.w;
    }

    const unsigned int* wbase = &sW[out_id][0];
#pragma unroll
    for (int g = 0; g < 2; ++g) {
      Frag bf[4];
#pragma unroll
      for (int j = 0; j < 4; ++j) {
        int nt = g * 4 + j;
        const unsigned int* pb = wbase + (nt * 16 + l15) * WPAD + half * 4;
        uint4 b0 = *(const uint4*)pb;
        uint4 b1 = *(const uint4*)(pb + 8);
        bf[j].u[0] = b0.x; bf[j].u[1] = b0.y; bf[j].u[2] = b0.z; bf[j].u[3] = b0.w;
        bf[j].u[4] = b1.x; bf[j].u[5] = b1.y; bf[j].u[6] = b1.z; bf[j].u[7] = b1.w;
      }
#pragma unroll
      for (int j = 0; j < 4; ++j) {
        acc[g * 4 + j] = __builtin_amdgcn_wmma_f32_16x16x32_bf16(
            false, af.v, false, bf[j].v, (short)0, acc[g * 4 + j], false, false);
      }
    }

    // Pipeline spec for the post-RA scheduler: batch DS reads ahead of WMMAs
    // so waits are partial (s_wait_dscnt N) instead of 0 before every WMMA.
#if __has_builtin(__builtin_amdgcn_sched_group_barrier)
    __builtin_amdgcn_sched_group_barrier(0x100, 10, 0);  // 10 DS reads (A + bf0..3)
    __builtin_amdgcn_sched_group_barrier(0x008,  4, 0);  //  4 WMMA
    __builtin_amdgcn_sched_group_barrier(0x100,  8, 0);  //  8 DS reads (bf4..7)
    __builtin_amdgcn_sched_group_barrier(0x008,  4, 0);  //  4 WMMA
#endif
  }

  // ---- epilogue: bias add + store (C/D layout: VGPR v -> row v / v+8)
  const float* bias = sel_w(out_id, bzm, bzv, bqm, bqv);
  float* op = out + (size_t)out_id * ((size_t)Mrows * Gdim);
#pragma unroll
  for (int nt = 0; nt < 8; ++nt) {
    int col  = nt * 16 + l15;
    float bv = bias[col];
#pragma unroll
    for (int v = 0; v < 8; ++v) {
      int row = m0 + ms * 16 + v + half * 8;
      if (row < Mrows) op[(size_t)row * Gdim + col] = acc[nt][v] + bv;
    }
  }
}

// ---------------- KL reduction (two-stage, deterministic) ----------------
__global__ __launch_bounds__(256) void kl_partial_kernel(
    const float* __restrict__ out, float* __restrict__ ws) {
  __shared__ float red[256];
  const size_t S = (size_t)Mrows * Gdim;
  float sum = 0.f;
  for (size_t i = (size_t)blockIdx.x * blockDim.x + threadIdx.x; i < S;
       i += (size_t)gridDim.x * blockDim.x) {
    float zm  = out[i];
    float zlv = out[S + i];
    float qm  = out[2 * S + i];
    float qlv = out[3 * S + i];
    float d = zm - qm;
    sum += qlv - zlv + (expf(zlv) + d * d) * expf(-qlv) - 1.0f;
  }
  red[threadIdx.x] = sum;
  __syncthreads();
  for (int s = 128; s > 0; s >>= 1) {
    if (threadIdx.x < s) red[threadIdx.x] += red[threadIdx.x + s];
    __syncthreads();
  }
  if (threadIdx.x == 0) ws[blockIdx.x] = red[0];
}

__global__ __launch_bounds__(256) void kl_final_kernel(
    const float* __restrict__ ws, float* __restrict__ out) {
  __shared__ float red[256];
  red[threadIdx.x] = ws[threadIdx.x];
  __syncthreads();
  for (int s = 128; s > 0; s >>= 1) {
    if (threadIdx.x < s) red[threadIdx.x] += red[threadIdx.x + s];
    __syncthreads();
  }
  if (threadIdx.x == 0) {
    const size_t S = (size_t)Mrows * Gdim;
    out[4 * S] = 0.5f * red[0] / (float)Mrows;
  }
}

extern "C" void kernel_launch(void* const* d_in, const int* in_sizes, int n_in,
                              void* d_out, int out_size, void* d_ws, size_t ws_size,
                              hipStream_t stream) {
  (void)in_sizes; (void)n_in; (void)out_size; (void)ws_size;
  const float* events   = (const float*)d_in[0];
  const float* contexts = (const float*)d_in[1];
  const float* Wzm = (const float*)d_in[2];
  const float* bzm = (const float*)d_in[3];
  const float* Wzv = (const float*)d_in[4];
  const float* bzv = (const float*)d_in[5];
  const float* Wqm = (const float*)d_in[6];
  const float* bqm = (const float*)d_in[7];
  const float* Wqv = (const float*)d_in[8];
  const float* bqv = (const float*)d_in[9];
  float* out = (float*)d_out;
  float* ws  = (float*)d_ws;   // needs 256 floats

  dim3 grid((Mrows + MTILE - 1) / MTILE);   // 256 blocks
  cvae_gemm_kernel<<<grid, NTHREADS, 0, stream>>>(
      events, contexts, Wzm, bzm, Wzv, bzv, Wqm, bqm, Wqv, bqv, out);
  kl_partial_kernel<<<256, 256, 0, stream>>>(out, ws);
  kl_final_kernel<<<1, 256, 0, stream>>>(ws, out);
}
